// PCT_14276471292458
// MI455X (gfx1250) — matrix operand order repaired and checked by reading, not compiled, not run on hardware
//
#include <hip/hip_runtime.h>

// ---------------------------------------------------------------------------
// PCT forward for MI455X (gfx1250): all dense matmuls via v_wmma_f32_16x16x32_bf16
// GEMM wave tile register-blocked to 32x64 (2x4 subtiles, 8 WMMAs / K-step).
// ---------------------------------------------------------------------------

typedef __attribute__((ext_vector_type(16))) __bf16       v16bf;
typedef __attribute__((ext_vector_type(8)))  float        v8f;
typedef __attribute__((ext_vector_type(8)))  unsigned int v8u;

__device__ __forceinline__ unsigned short f32_to_bf16(float f) {
  unsigned int u = __float_as_uint(f);
  u += 0x7FFFu + ((u >> 16) & 1u);           // round-to-nearest-even
  return (unsigned short)(u >> 16);
}
__device__ __forceinline__ float bf16_to_f32(unsigned short h) {
  return __uint_as_float(((unsigned int)h) << 16);
}

// ---------------------------------------------------------------------------
// Generic batched GEMM:  C[b] = act(affine(A[b] * B[b])) (+/- residual)
//   A: bf16 [M,K] row-major (lda), batch stride aBS (elements)
//   B: bf16 [K,N] row-major (ldb), batch stride bBS
//   One wave computes a 32x64 output block (2 M-subtiles x 4 N-subtiles),
//   K stepped by 32; A fragments reused across 4 N-subtiles, B across 2 M.
// Fragment packing follows CDNA5 ISA 7.12.2 (wave32):
//   A 16x32: lanes 0-15 row m, K pairs {2r,2r+1} r<4 then {16+2r}, lanes 16-31 +8
//   B 32x16: lanes 0-15 col n, same K order as A (pairs {2r} r<4 then {16+2r}), lanes 16-31 +8
//   C/D   : col = lane&15, row = r + 8*(lane>>4)
// Requires: M % 32 == 0, K % 32 == 0 (N ragged handled by whole-wave exit).
// ---------------------------------------------------------------------------
__global__ __launch_bounds__(256) void wmma_gemm(
    const unsigned short* __restrict__ A, long aBS, int lda,
    const unsigned short* __restrict__ Bm, long bBS, int ldb,
    int M, int K, int NN,
    const float* __restrict__ scale, const float* __restrict__ shift,
    const unsigned short* __restrict__ res, long rBS, int ldr, int resMode, // 0 none,1 out+=res,2 out=res-out
    float* __restrict__ outF, long fBS, int ldf,
    unsigned short* __restrict__ outH, long hBS, int ldh,
    int act)                                                                // 0 none,1 relu,2 leaky0.2
{
  const int lane   = threadIdx.x;             // 0..31
  const int wave   = threadIdx.y;             // 0..7
  const int nchunk = blockIdx.x * 8 + wave;   // 64-wide output chunk
  const int mchunk = blockIdx.y;              // 32-tall output chunk
  const int b      = blockIdx.z;
  if (nchunk * 64 >= NN) return;              // uniform across wave -> EXEC stays full

  const unsigned short* Ab = A  + (size_t)b * aBS;
  const unsigned short* Bb = Bm + (size_t)b * bBS;
  const int h = lane >> 4;                    // lane half
  const int m = lane & 15;
  const int arow0 = mchunk * 32 + m;          // A row, M-subtile 0
  const int arow1 = arow0 + 16;               // A row, M-subtile 1
  const int bcol0 = nchunk * 64 + m;          // B col base for this lane

  v8f acc[2][4] = {};

  for (int kb = 0; kb < K; kb += 32) {
    v8u a0, a1;
#pragma unroll
    for (int r = 0; r < 8; ++r) {
      const int ak = ((r < 4) ? (2 * r) : (16 + 2 * (r - 4))) + h * 8;
      a0[r] = *(const unsigned int*)(Ab + (size_t)arow0 * lda + kb + ak);
      a1[r] = *(const unsigned int*)(Ab + (size_t)arow1 * lda + kb + ak);
    }
    const v16bf av0 = __builtin_bit_cast(v16bf, a0);
    const v16bf av1 = __builtin_bit_cast(v16bf, a1);
#pragma unroll
    for (int j = 0; j < 4; ++j) {
      const int bc = bcol0 + j * 16;
      v8u braw;
#pragma unroll
      for (int r = 0; r < 8; ++r) {
        const int bk = ((r < 4) ? (2 * r) : (16 + 2 * (r - 4))) + h * 8;
        unsigned int lo = Bb[(size_t)(kb + bk)     * ldb + bc];
        unsigned int hi = Bb[(size_t)(kb + bk + 1) * ldb + bc];
        braw[r] = lo | (hi << 16);
      }
      const v16bf bv = __builtin_bit_cast(v16bf, braw);
      acc[0][j] = __builtin_amdgcn_wmma_f32_16x16x32_bf16(false, av0, false, bv,
                                                          (short)0, acc[0][j], false, false);
      acc[1][j] = __builtin_amdgcn_wmma_f32_16x16x32_bf16(false, av1, false, bv,
                                                          (short)0, acc[1][j], false, false);
    }
  }

#pragma unroll
  for (int mi = 0; mi < 2; ++mi) {
#pragma unroll
    for (int r = 0; r < 8; ++r) {
      const int row = mchunk * 32 + mi * 16 + r + h * 8;
      const float sc = scale ? scale[row] : 1.0f;
      const float sh = shift ? shift[row] : 0.0f;
#pragma unroll
      for (int j = 0; j < 4; ++j) {
        const int col = nchunk * 64 + j * 16 + m;
        float v = acc[mi][j][r];
        if (scale)      v = v * sc + sh;
        else if (shift) v = v + sh;
        if (act == 1)      v = v > 0.0f ? v : 0.0f;
        else if (act == 2) v = v > 0.0f ? v : 0.2f * v;
        if (resMode) {
          float rv = bf16_to_f32(res[(size_t)b * rBS + (size_t)row * ldr + col]);
          v = (resMode == 1) ? (v + rv) : (rv - v);
        }
        if (outF) outF[(size_t)b * fBS + (size_t)row * ldf + col] = v;
        if (outH) outH[(size_t)b * hBS + (size_t)row * ldh + col] = f32_to_bf16(v);
      }
    }
  }
}

// --- BN fold: s = g*rsqrt(v+eps); t = (extra - m)*s + b -----------------------
__global__ void bn_prep_kernel(const float* __restrict__ g, const float* __restrict__ bb,
                               const float* __restrict__ mm, const float* __restrict__ vv,
                               const float* __restrict__ extra, int n,
                               float* __restrict__ s, float* __restrict__ t)
{
  int i = blockIdx.x * blockDim.x + threadIdx.x;
  if (i >= n) return;
  float sc = g[i] * rsqrtf(vv[i] + 1e-5f);
  float base = (extra ? extra[i] : 0.0f) - mm[i];
  s[i] = sc;
  t[i] = base * sc + bb[i];
}

__global__ void cvt_bf16_kernel(const float* __restrict__ in, unsigned short* __restrict__ out, long n)
{
  long i = (long)blockIdx.x * blockDim.x + threadIdx.x;
  if (i < n) out[i] = f32_to_bf16(in[i]);
}

// --- conv1: [B,3,N] x W[64,3] + BN + relu -> bf16 [B,64,N] -------------------
__global__ void conv1_kernel(const float* __restrict__ coords, const float* __restrict__ w,
                             const float* __restrict__ sc, const float* __restrict__ sh,
                             unsigned short* __restrict__ out, int Bn, int N)
{
  int idx = blockIdx.x * blockDim.x + threadIdx.x;
  if (idx >= Bn * N) return;
  int b = idx / N, n = idx % N;
  const float* cb = coords + (size_t)b * 3 * N;
  float x0 = cb[n], x1 = cb[N + n], x2 = cb[2 * N + n];
  for (int o = 0; o < 64; ++o) {
    float v = w[o * 3] * x0 + w[o * 3 + 1] * x1 + w[o * 3 + 2] * x2;
    v = v * sc[o] + sh[o];
    v = v > 0.0f ? v : 0.0f;
    out[((size_t)b * 64 + o) * N + n] = f32_to_bf16(v);
  }
}

// --- farthest point sampling (one block per batch) ---------------------------
// xyz value = base[b*bOff + n*sN + c*sC]; matches jnp argmax tie->lowest index
__global__ __launch_bounds__(256) void fps_kernel(
    const float* __restrict__ xyz, long bOff, int sN, int sC,
    int N, int npoint, int* __restrict__ out)
{
  __shared__ float dist[4096];
  __shared__ float rv[256];
  __shared__ int   ri[256];
  __shared__ float cen[3];
  int b = blockIdx.x, t = threadIdx.x;
  const float* base = xyz + (size_t)b * bOff;
  for (int n = t; n < N; n += 256) dist[n] = 1e10f;
  __syncthreads();
  int far = 0;
  for (int it = 0; it < npoint; ++it) {
    if (t == 0) {
      out[b * npoint + it] = far;
      cen[0] = base[(size_t)far * sN + 0 * sC];
      cen[1] = base[(size_t)far * sN + 1 * sC];
      cen[2] = base[(size_t)far * sN + 2 * sC];
    }
    __syncthreads();
    float bv = -1.0f; int bi = 0x7fffffff;
    for (int n = t; n < N; n += 256) {
      float dx = base[(size_t)n * sN + 0 * sC] - cen[0];
      float dy = base[(size_t)n * sN + 1 * sC] - cen[1];
      float dz = base[(size_t)n * sN + 2 * sC] - cen[2];
      float d  = dx * dx + dy * dy + dz * dz;
      float od = dist[n];
      d = d < od ? d : od;
      dist[n] = d;
      if (d > bv || (d == bv && n < bi)) { bv = d; bi = n; }
    }
    rv[t] = bv; ri[t] = bi;
    __syncthreads();
    for (int s2 = 128; s2 > 0; s2 >>= 1) {
      if (t < s2) {
        if (rv[t + s2] > rv[t] || (rv[t + s2] == rv[t] && ri[t + s2] < ri[t])) {
          rv[t] = rv[t + s2]; ri[t] = ri[t + s2];
        }
      }
      __syncthreads();
    }
    far = ri[0];
    __syncthreads();
  }
}

// --- kNN: 32 smallest distances (one block per query point) ------------------
__global__ __launch_bounds__(128) void knn_kernel(
    const float* __restrict__ xyz, long bOff, int sN, int sC, int N,
    const int* __restrict__ fps, int S, int Ksel, int* __restrict__ out)
{
  __shared__ float d[4096];
  __shared__ float rv[128];
  __shared__ int   ri[128];
  int bs = blockIdx.x, t = threadIdx.x;
  int b = bs / S, s = bs % S;
  const float* base = xyz + (size_t)b * bOff;
  int ci = fps[b * S + s];
  float cx = base[(size_t)ci * sN + 0 * sC];
  float cy = base[(size_t)ci * sN + 1 * sC];
  float cz = base[(size_t)ci * sN + 2 * sC];
  for (int n = t; n < N; n += 128) {
    float dx = base[(size_t)n * sN + 0 * sC] - cx;
    float dy = base[(size_t)n * sN + 1 * sC] - cy;
    float dz = base[(size_t)n * sN + 2 * sC] - cz;
    d[n] = dx * dx + dy * dy + dz * dz;
  }
  __syncthreads();
  for (int k = 0; k < Ksel; ++k) {
    float bv = 3.4e38f; int bi = 0x7fffffff;
    for (int n = t; n < N; n += 128)
      if (d[n] < bv || (d[n] == bv && n < bi)) { bv = d[n]; bi = n; }
    rv[t] = bv; ri[t] = bi;
    __syncthreads();
    for (int s2 = 64; s2 > 0; s2 >>= 1) {
      if (t < s2) {
        if (rv[t + s2] < rv[t] || (rv[t + s2] == rv[t] && ri[t + s2] < ri[t])) {
          rv[t] = rv[t + s2]; ri[t] = ri[t + s2];
        }
      }
      __syncthreads();
    }
    if (t == 0) { out[((size_t)b * S + s) * Ksel + k] = ri[0]; d[ri[0]] = 3.4e38f; }
    __syncthreads();
  }
}

// --- gather sampled xyz: out[b][s][c] = coords[b][c][fps] --------------------
__global__ void gatherxyz_kernel(const float* __restrict__ coords, int N,
                                 const int* __restrict__ fps, int total,
                                 float* __restrict__ out)
{
  int idx = blockIdx.x * blockDim.x + threadIdx.x;   // b*S + s
  if (idx >= total) return;
  int S = total / 16;
  int b = idx / S;
  int n = fps[idx];
  const float* cb = coords + (size_t)b * 3 * N;
  out[(size_t)idx * 3 + 0] = cb[n];
  out[(size_t)idx * 3 + 1] = cb[N + n];
  out[(size_t)idx * 3 + 2] = cb[2 * N + n];
}

// --- grouping: grouped[0..C-1]=g-c, grouped[C..2C-1]=c, bf16 -----------------
// pts value = pts[b*bStr + c*cStr + n]
__global__ void group_kernel(const unsigned short* __restrict__ pts, long bStr, long cStr,
                             const int* __restrict__ fps, const int* __restrict__ knn,
                             int S, int Ksel, int C,
                             unsigned short* __restrict__ out, long cols)
{
  long col = (long)blockIdx.x * blockDim.x + threadIdx.x;
  if (col >= cols) return;
  int k  = (int)(col % Ksel);
  long bs = col / Ksel;
  int s  = (int)(bs % S);
  int b  = (int)(bs / S);
  int cen = fps[b * S + s];
  int gn  = knn[((size_t)b * S + s) * Ksel + k];
  (void)k;
  for (int c = 0; c < C; ++c) {
    float cv = bf16_to_f32(pts[(size_t)b * bStr + (size_t)c * cStr + cen]);
    float gv = bf16_to_f32(pts[(size_t)b * bStr + (size_t)c * cStr + gn]);
    out[(size_t)c * cols + col]       = f32_to_bf16(gv - cv);
    out[(size_t)(C + c) * cols + col] = f32_to_bf16(cv);
  }
}

// --- max over groups of `grp` columns ---------------------------------------
__global__ void maxgroup_kernel(const unsigned short* __restrict__ in, long cols, int grp,
                                unsigned short* __restrict__ out, long outCols, int C)
{
  long idx = (long)blockIdx.x * blockDim.x + threadIdx.x;
  if (idx >= (long)C * outCols) return;
  int  o    = (int)(idx / outCols);
  long gcol = idx % outCols;
  const unsigned short* p = in + (size_t)o * cols + gcol * grp;
  float mx = -3.4e38f;
  for (int k = 0; k < grp; ++k) {
    float v = bf16_to_f32(p[k]);
    mx = v > mx ? v : mx;
  }
  out[idx] = f32_to_bf16(mx);
}

// --- bf16 transpose per batch: out[b][c][r] = in[b][r][c] --------------------
__global__ void transpose_kernel(const unsigned short* __restrict__ in, int R, int Cc,
                                 long bStr, unsigned short* __restrict__ out, long obStr,
                                 int total)
{
  int idx = blockIdx.x * blockDim.x + threadIdx.x;
  if (idx >= total) return;
  int c = idx % Cc;
  int r = (idx / Cc) % R;
  int b = idx / (Cc * R);
  out[(size_t)b * obStr + (size_t)c * R + r] = in[(size_t)b * bStr + (size_t)r * Cc + c];
}

// --- row softmax over n=256 (block per row) ----------------------------------
__global__ __launch_bounds__(256) void softmax_row(float* __restrict__ data, int n)
{
  __shared__ float red[256];
  int row = blockIdx.x, t = threadIdx.x;
  float* p = data + (size_t)row * n;
  float v = p[t];
  red[t] = v; __syncthreads();
  for (int s = 128; s > 0; s >>= 1) { if (t < s) red[t] = fmaxf(red[t], red[t + s]); __syncthreads(); }
  float mx = red[0]; __syncthreads();
  float e = __expf(v - mx);
  red[t] = e; __syncthreads();
  for (int s = 128; s > 0; s >>= 1) { if (t < s) red[t] += red[t + s]; __syncthreads(); }
  p[t] = e / red[0];
}

// --- column normalize attn/(1e-9+colsum) -> bf16 (block per batch) ----------
__global__ __launch_bounds__(256) void colnorm_kernel(const float* __restrict__ attn,
                                                      unsigned short* __restrict__ outh, int n)
{
  int b = blockIdx.x, t = threadIdx.x;     // column m = t
  const float* p = attn + (size_t)b * n * n;
  unsigned short* o = outh + (size_t)b * n * n;
  float s = 0.0f;
  for (int r = 0; r < n; ++r) s += p[(size_t)r * n + t];
  float inv = 1.0f / (1e-9f + s);
  for (int r = 0; r < n; ++r) o[(size_t)r * n + t] = f32_to_bf16(p[(size_t)r * n + t] * inv);
}

// --- copy l2 into fuse-input rows 1024..1279 ---------------------------------
__global__ void copyl2_kernel(const unsigned short* __restrict__ l2h,
                              unsigned short* __restrict__ feat, int total)
{
  int idx = blockIdx.x * blockDim.x + threadIdx.x;
  if (idx >= total) return;
  int j = idx & 255;
  int c = (idx >> 8) & 255;
  int b = idx >> 16;
  feat[(size_t)b * 1280 * 256 + (size_t)(1024 + c) * 256 + j] =
      l2h[(size_t)c * 4096 + (size_t)b * 256 + j];
}

// --- pooled = [max over N | mean over N] -------------------------------------
__global__ void pool_kernel(const float* __restrict__ feature, float* __restrict__ pooled, int total)
{
  int idx = blockIdx.x * blockDim.x + threadIdx.x;
  if (idx >= total) return;
  int b = idx / 2048, ch = idx % 2048;
  const float* p = feature + ((size_t)b * 2048 + ch) * 256;
  float mx = -3.4e38f, sm = 0.0f;
  for (int j = 0; j < 256; ++j) { float v = p[j]; mx = fmaxf(mx, v); sm += v; }
  pooled[(size_t)b * 4096 + ch]        = mx;
  pooled[(size_t)b * 4096 + 2048 + ch] = sm * (1.0f / 256.0f);
}

// ===========================================================================

static inline void gemm(hipStream_t st,
                        const unsigned short* A, long aBS, int lda,
                        const unsigned short* Bm, long bBS, int ldb,
                        int M, int K, int NN, int batches,
                        const float* scale, const float* shift,
                        const unsigned short* res, long rBS, int ldr, int resMode,
                        float* outF, long fBS, int ldf,
                        unsigned short* outH, long hBS, int ldh, int act)
{
  dim3 blk(32, 8, 1);
  int nChunks = (NN + 63) / 64;                 // 64-wide wave chunks
  dim3 grd((nChunks + 7) / 8, (M + 31) / 32, batches);
  wmma_gemm<<<grd, blk, 0, st>>>(A, aBS, lda, Bm, bBS, ldb, M, K, NN,
                                 scale, shift, res, rBS, ldr, resMode,
                                 outF, fBS, ldf, outH, hBS, ldh, act);
}

extern "C" void kernel_launch(void* const* d_in, const int* in_sizes, int n_in,
                              void* d_out, int out_size, void* d_ws, size_t ws_size,
                              hipStream_t stream)
{
  (void)in_sizes; (void)n_in; (void)out_size; (void)ws_size;
  const int B = 16, N0 = 4096, S1 = 512, S2 = 256, KS = 32;

  // ---- input indices (setup_inputs dict insertion order, depth-first) ----
  const float* coords = (const float*)d_in[0];
  const float* xyz_o  = (const float*)d_in[1];
  const float* center = (const float*)d_in[2];
  const float* conv1_w = (const float*)d_in[3];
  auto F = [&](int i) { return (const float*)d_in[i]; };

  // ---- workspace bump allocator (deterministic) ----
  char* ws = (char*)d_ws;
  size_t off = 0;
  auto alloc = [&](size_t bytes) -> char* {
    char* p = ws + off;
    off = (off + bytes + 255) & ~(size_t)255;
    return p;
  };
  unsigned short* BIG0 = (unsigned short*)alloc(67108864);   // 64 MiB ping
  unsigned short* BIG1 = (unsigned short*)alloc(67108864);   // 64 MiB pong
  unsigned short* x1h  = (unsigned short*)alloc((size_t)B * 64 * N0 * 2);
  unsigned short* x2h  = (unsigned short*)alloc((size_t)B * 64 * N0 * 2);
  unsigned short* l1h  = (unsigned short*)alloc((size_t)128 * B * S1 * 2);
  unsigned short* l2h  = (unsigned short*)alloc((size_t)256 * B * S2 * 2);
  unsigned short* feath= (unsigned short*)alloc((size_t)B * 1280 * 256 * 2);
  // weights (bf16)
  unsigned short* w_conv2 = (unsigned short*)alloc(64 * 64 * 2);
  unsigned short* w_l0a   = (unsigned short*)alloc(128 * 128 * 2);
  unsigned short* w_l0b   = (unsigned short*)alloc(128 * 128 * 2);
  unsigned short* w_l1a   = (unsigned short*)alloc(256 * 256 * 2);
  unsigned short* w_l1b   = (unsigned short*)alloc(256 * 256 * 2);
  unsigned short* w_att1  = (unsigned short*)alloc(256 * 256 * 2);
  unsigned short* w_att2  = (unsigned short*)alloc(256 * 256 * 2);
  unsigned short* w_qk[4], *w_v[4], *w_t[4];
  for (int i = 0; i < 4; ++i) {
    w_qk[i] = (unsigned short*)alloc(64 * 256 * 2);
    w_v[i]  = (unsigned short*)alloc(256 * 256 * 2);
    w_t[i]  = (unsigned short*)alloc(256 * 256 * 2);
  }
  unsigned short* w_fuse = (unsigned short*)alloc((size_t)2048 * 1280 * 2);
  // folded BN scale/shift
  auto allocST = [&](int n, float*& s, float*& t) {
    s = (float*)alloc((size_t)n * 4);
    t = (float*)alloc((size_t)n * 4);
  };
  float *bn1_s, *bn1_t, *bn2_s, *bn2_t;
  float *l0a_s, *l0a_t, *l0b_s, *l0b_t, *l1a_s, *l1a_t, *l1b_s, *l1b_t;
  float *a1_s, *a1_t, *a2_s, *a2_t, *fz_s, *fz_t;
  float *sa_s[4], *sa_t[4];
  allocST(64, bn1_s, bn1_t);   allocST(64, bn2_s, bn2_t);
  allocST(128, l0a_s, l0a_t);  allocST(128, l0b_s, l0b_t);
  allocST(256, l1a_s, l1a_t);  allocST(256, l1b_s, l1b_t);
  allocST(256, a1_s, a1_t);    allocST(256, a2_s, a2_t);
  for (int i = 0; i < 4; ++i) allocST(256, sa_s[i], sa_t[i]);
  allocST(2048, fz_s, fz_t);
  // sampling buffers
  int*   fps1   = (int*)alloc((size_t)B * S1 * 4);
  int*   knn1   = (int*)alloc((size_t)B * S1 * KS * 4);
  float* l1xyz  = (float*)alloc((size_t)B * S1 * 3 * 4);
  int*   fps2   = (int*)alloc((size_t)B * S2 * 4);
  int*   knn2   = (int*)alloc((size_t)B * S2 * KS * 4);
  // attention intermediates alias BIG0 (free after local-op stages)
  unsigned short* y0h   = (unsigned short*)((char*)BIG0 + 0);         // 2 MiB
  unsigned short* y1h   = (unsigned short*)((char*)BIG0 + 2097152);   // 2 MiB
  unsigned short* kh    = (unsigned short*)((char*)BIG0 + 4194304);   // 512 KiB
  unsigned short* ktr   = (unsigned short*)((char*)BIG0 + 4718592);   // 512 KiB
  float*          logit = (float*)((char*)BIG0 + 5242880);            // 4 MiB
  unsigned short* attnh = (unsigned short*)((char*)BIG0 + 9437184);   // 2 MiB
  unsigned short* vh    = (unsigned short*)((char*)BIG0 + 11534336);  // 2 MiB
  unsigned short* zh    = (unsigned short*)((char*)BIG0 + 13631488);  // 2 MiB

  // ---- 1) fold BN params ----
  auto bnprep = [&](int gi, const float* extra, int n, float* s, float* t) {
    bn_prep_kernel<<<(n + 255) / 256, 256, 0, stream>>>(F(gi), F(gi + 1), F(gi + 2), F(gi + 3),
                                                        extra, n, s, t);
  };
  bnprep(4, nullptr, 64, bn1_s, bn1_t);
  bnprep(9, nullptr, 64, bn2_s, bn2_t);
  bnprep(14, nullptr, 128, l0a_s, l0a_t);
  bnprep(19, nullptr, 128, l0b_s, l0b_t);
  bnprep(24, nullptr, 256, l1a_s, l1a_t);
  bnprep(29, nullptr, 256, l1b_s, l1b_t);
  bnprep(34, nullptr, 256, a1_s, a1_t);
  bnprep(39, nullptr, 256, a2_s, a2_t);
  for (int i = 0; i < 4; ++i) {
    int base = 43 + i * 9;                     // qk_w, v_w, v_b, t_w, t_b, g,b,m,v
    bnprep(base + 5, F(base + 4), 256, sa_s[i], sa_t[i]);  // fold t_b into BN shift
  }
  bnprep(80, nullptr, 2048, fz_s, fz_t);

  // ---- 2) convert weights to bf16 ----
  auto cvt = [&](int i, unsigned short* dst, long n) {
    cvt_bf16_kernel<<<(int)((n + 255) / 256), 256, 0, stream>>>(F(i), dst, n);
  };
  cvt(8,  w_conv2, 64 * 64);
  cvt(13, w_l0a, 128 * 128);  cvt(18, w_l0b, 128 * 128);
  cvt(23, w_l1a, 256 * 256);  cvt(28, w_l1b, 256 * 256);
  cvt(33, w_att1, 256 * 256); cvt(38, w_att2, 256 * 256);
  for (int i = 0; i < 4; ++i) {
    int base = 43 + i * 9;
    cvt(base + 0, w_qk[i], 64 * 256);
    cvt(base + 1, w_v[i], 256 * 256);
    cvt(base + 3, w_t[i], 256 * 256);
  }
  cvt(79, w_fuse, (long)2048 * 1280);

  // ---- 3) conv1 + bn1 + relu (K=3, VALU) -> x1h ----
  conv1_kernel<<<(B * N0 + 255) / 256, 256, 0, stream>>>(coords, conv1_w, bn1_s, bn1_t, x1h, B, N0);

  // ---- 4) conv2 + bn2 + relu (WMMA) -> x2h ----
  gemm(stream, w_conv2, 0, 64, x1h, (long)64 * N0, N0, 64, 64, N0, B,
       bn2_s, bn2_t, nullptr, 0, 0, 0, nullptr, 0, 0, x2h, (long)64 * N0, N0, 1);

  // ---- 5) stage-1 sampling/grouping ----
  fps_kernel<<<B, 256, 0, stream>>>(coords, (long)3 * N0, 1, N0, N0, S1, fps1);
  knn_kernel<<<B * S1, 128, 0, stream>>>(coords, (long)3 * N0, 1, N0, N0, fps1, S1, KS, knn1);
  gatherxyz_kernel<<<(B * S1 + 255) / 256, 256, 0, stream>>>(coords, N0, fps1, B * S1, l1xyz);
  const long cols1 = (long)B * S1 * KS;                 // 262144
  group_kernel<<<(int)((cols1 + 255) / 256), 256, 0, stream>>>(
      x2h, (long)64 * N0, N0, fps1, knn1, S1, KS, 64, BIG0, cols1);

  // ---- 6) local op 0: two 128x128 WMMA GEMMs + group max ----
  gemm(stream, w_l0a, 0, 128, BIG0, 0, (int)cols1, 128, 128, (int)cols1, 1,
       l0a_s, l0a_t, nullptr, 0, 0, 0, nullptr, 0, 0, BIG1, 0, (int)cols1, 1);
  gemm(stream, w_l0b, 0, 128, BIG1, 0, (int)cols1, 128, 128, (int)cols1, 1,
       l0b_s, l0b_t, nullptr, 0, 0, 0, nullptr, 0, 0, BIG0, 0, (int)cols1, 1);
  maxgroup_kernel<<<(int)(((long)128 * B * S1 + 255) / 256), 256, 0, stream>>>(
      BIG0, cols1, KS, l1h, (long)B * S1, 128);

  // ---- 7) stage-2 sampling/grouping (on l1) ----
  fps_kernel<<<B, 256, 0, stream>>>(l1xyz, (long)S1 * 3, 3, 1, S1, S2, fps2);
  knn_kernel<<<B * S2, 128, 0, stream>>>(l1xyz, (long)S1 * 3, 3, 1, S1, fps2, S2, KS, knn2);
  const long cols2 = (long)B * S2 * KS;                 // 131072
  group_kernel<<<(int)((cols2 + 255) / 256), 256, 0, stream>>>(
      l1h, (long)S1, (long)B * S1, fps2, knn2, S2, KS, 128, BIG0, cols2);

  // ---- 8) local op 1: two 256x256 WMMA GEMMs + group max -> l2h ----
  gemm(stream, w_l1a, 0, 256, BIG0, 0, (int)cols2, 256, 256, (int)cols2, 1,
       l1a_s, l1a_t, nullptr, 0, 0, 0, nullptr, 0, 0, BIG1, 0, (int)cols2, 1);
  gemm(stream, w_l1b, 0, 256, BIG1, 0, (int)cols2, 256, 256, (int)cols2, 1,
       l1b_s, l1b_t, nullptr, 0, 0, 0, nullptr, 0, 0, BIG0, 0, (int)cols2, 1);
  maxgroup_kernel<<<(int)(((long)256 * B * S2 + 255) / 256), 256, 0, stream>>>(
      BIG0, cols2, KS, l2h, (long)B * S2, 256);

  // ---- 9) attention front convs (per-batch 256x256 WMMA) ----
  // l2h per-batch B matrix: base + b*256, ldb = B*S2 = 4096
  gemm(stream, w_att1, 0, 256, l2h, 256, B * S2, 256, 256, 256, B,
       a1_s, a1_t, nullptr, 0, 0, 0, nullptr, 0, 0, y0h, 65536, 256, 1);
  gemm(stream, w_att2, 0, 256, y0h, 65536, 256, 256, 256, 256, B,
       a2_s, a2_t, nullptr, 0, 0, 0, nullptr, 0, 0, y1h, 65536, 256, 1);

  // ---- 10) four offset-attention layers ----
  for (int i = 0; i < 4; ++i) {
    int base = 43 + i * 9;
    const unsigned short* ycur = (i == 0) ? y1h : (feath + (size_t)(i - 1) * 65536);
    long yBS = (i == 0) ? 65536L : 327680L;   // feat batch stride = 1280*256
    // K = qk_w * y  -> kh [B,64,256]
    gemm(stream, w_qk[i], 0, 256, ycur, yBS, 256, 64, 256, 256, B,
         nullptr, nullptr, nullptr, 0, 0, 0, nullptr, 0, 0, kh, 16384, 256, 0);
    // ktr[b][n][c] = kh[b][c][n]
    transpose_kernel<<<(B * 64 * 256 + 255) / 256, 256, 0, stream>>>(
        kh, 64, 256, 16384, ktr, 16384, B * 64 * 256);
    // S = Kt * K -> logits f32 [B,256,256]
    gemm(stream, ktr, 16384, 64, kh, 16384, 256, 256, 64, 256, B,
         nullptr, nullptr, nullptr, 0, 0, 0, logit, 65536, 256, nullptr, 0, 0, 0);
    // softmax rows, then column-normalize -> bf16 attn
    softmax_row<<<B * 256, 256, 0, stream>>>(logit, 256);
    colnorm_kernel<<<B, 256, 0, stream>>>(logit, attnh, 256);
    // V = v_w * y + v_b -> vh
    gemm(stream, w_v[i], 0, 256, ycur, yBS, 256, 256, 256, 256, B,
         nullptr, F(base + 2), nullptr, 0, 0, 0, nullptr, 0, 0, vh, 65536, 256, 0);
    // z = y - V*attn  (residual mode 2)
    gemm(stream, vh, 65536, 256, attnh, 65536, 256, 256, 256, 256, B,
         nullptr, nullptr, ycur, yBS, 256, 2, nullptr, 0, 0, zh, 65536, 256, 0);
    // y_next = y + relu(bn(t_w*z + t_b))  -> feat slot i (rows i*256..)
    gemm(stream, w_t[i], 0, 256, zh, 65536, 256, 256, 256, 256, B,
         sa_s[i], sa_t[i], ycur, yBS, 256, 1, nullptr, 0, 0,
         feath + (size_t)i * 65536, 327680, 256, 1);
  }

  // ---- 11) append l2 as fuse-input rows 1024..1279 ----
  copyl2_kernel<<<(B * 256 * 256 + 255) / 256, 256, 0, stream>>>(l2h, feath, B * 256 * 256);

  // ---- 12) fuse GEMM 2048x1280 + BN + leaky(0.2) -> feature (f32, to d_out) ----
  float* out_pooled  = (float*)d_out;                      // [B,4096]
  float* out_feature = out_pooled + (size_t)B * 4096;      // [B,2048,256]
  gemm(stream, w_fuse, 0, 1280, feath, 327680, 256, 2048, 1280, 256, B,
       fz_s, fz_t, nullptr, 0, 0, 0, out_feature, (long)2048 * 256, 256, nullptr, 0, 0, 2);

  // ---- 13) pooled = [max | mean] ----
  pool_kernel<<<(B * 2048 + 255) / 256, 256, 0, stream>>>(out_feature, out_pooled, B * 2048);

  // ---- 14) pass-through outputs ----
  float* out_xyzo   = out_feature + (size_t)B * 2048 * 256;
  float* out_center = out_xyzo + (size_t)B * N0 * 3;
  (void)hipMemcpyAsync(out_xyzo, xyz_o, (size_t)B * N0 * 3 * 4, hipMemcpyDeviceToDevice, stream);
  (void)hipMemcpyAsync(out_center, center, (size_t)B * 3 * 4, hipMemcpyDeviceToDevice, stream);
}
